// HyperGRUCell_57578331570312
// MI455X (gfx1250) — matrix operand order repaired and checked by reading, not compiled, not run on hardware
//
#include <hip/hip_runtime.h>
#include <stdint.h>

typedef __attribute__((ext_vector_type(16))) __bf16 v16bf;
typedef __attribute__((ext_vector_type(8)))  float  v8f;
typedef __attribute__((ext_vector_type(4)))  __bf16 bf16x4;

#define B_DIM 4096
#define K_DIM 1024
#define N_DIM 1024

#define TM 128
#define TN 128
#define TK 32
#define PITCH 40      // bf16 elements per LDS row (32 data + 8 pad)

// ---------------------------------------------------------------------------
// 1) f32 -> bf16 cast (vectorized)
// ---------------------------------------------------------------------------
__global__ void cast_f32_bf16(const float* __restrict__ src,
                              __bf16* __restrict__ dst, int n4)
{
  int i = blockIdx.x * blockDim.x + threadIdx.x;
  if (i < n4) {
    float4 v = ((const float4*)src)[i];
    bf16x4 o;
    o.x = (__bf16)v.x; o.y = (__bf16)v.y; o.z = (__bf16)v.z; o.w = (__bf16)v.w;
    ((bf16x4*)dst)[i] = o;
  }
}

// ---------------------------------------------------------------------------
// 2) W[k][n] f32  ->  Wt[n][k] bf16  (LDS-tiled transpose, coalesced both ways)
// ---------------------------------------------------------------------------
__global__ void transpose_cast(const float* __restrict__ W,
                               __bf16* __restrict__ Wt, int K, int N)
{
  __shared__ float tile[32][33];
  const int bx = blockIdx.x * 32;   // n block
  const int by = blockIdx.y * 32;   // k block
  const int tx = threadIdx.x & 31;
  const int ty = threadIdx.x >> 5;  // 0..7
  #pragma unroll
  for (int i = 0; i < 32; i += 8)
    tile[ty + i][tx] = W[(size_t)(by + ty + i) * N + (bx + tx)];
  __syncthreads();
  #pragma unroll
  for (int i = 0; i < 32; i += 8)
    Wt[(size_t)(bx + ty + i) * K + (by + tx)] = (__bf16)tile[tx][ty + i];
}

// ---------------------------------------------------------------------------
// 3) GEMM: C(MxN) = A(MxK, bf16 row-major) * Bt(NxK, bf16 row-major, i.e. W^T)
//    128x128 block tile, 8 waves of 2x4, per wave 64x32 = 4x2 WMMA fragments.
//    LDS double-buffered, filled by global_load_async_to_lds_b128 (ASYNCcnt).
// ---------------------------------------------------------------------------
__device__ __forceinline__ v16bf ld_frag2(const __bf16* p0, const __bf16* p1)
{
  v16bf r;
  __builtin_memcpy(&r, p0, 16);
  __builtin_memcpy((char*)&r + 16, p1, 16);
  return r;
}

__global__ __launch_bounds__(256)
void gemm_bf16_wmma(const __bf16* __restrict__ A,
                    const __bf16* __restrict__ Bt,
                    float* __restrict__ C, int M, int N, int K)
{
  __shared__ __bf16 Al[2][TM * PITCH];
  __shared__ __bf16 Bl[2][TN * PITCH];

  const int t    = threadIdx.x;
  const int lane = t & 31;
  const int wave = t >> 5;          // 0..7
  const int wm   = wave & 1;        // 2 waves along M
  const int wn   = wave >> 1;       // 4 waves along N
  const int m0   = blockIdx.y * TM;
  const int n0   = blockIdx.x * TN;

  // async-copy work split: 128 rows x 4 x 16B segments per matrix per tile,
  // 256 threads -> 2 instructions per thread per matrix (4 per tile).
  const int rowA = t >> 2;          // 0..63
  const int seg  = t & 3;           // which 16B segment of the 64B k-slab

  auto issue_tile = [&](int kt, int buf) {
    #pragma unroll
    for (int q = 0; q < 2; ++q) {
      const int row = rowA + q * 64;
      unsigned long long ga =
          (unsigned long long)(A + (size_t)(m0 + row) * K + kt * TK + seg * 8);
      unsigned la = (unsigned)(unsigned long long)&Al[buf][row * PITCH + seg * 8];
      asm volatile("global_load_async_to_lds_b128 %0, %1, off"
                   :: "v"(la), "v"(ga) : "memory");
      unsigned long long gb =
          (unsigned long long)(Bt + (size_t)(n0 + row) * K + kt * TK + seg * 8);
      unsigned lb = (unsigned)(unsigned long long)&Bl[buf][row * PITCH + seg * 8];
      asm volatile("global_load_async_to_lds_b128 %0, %1, off"
                   :: "v"(lb), "v"(gb) : "memory");
    }
  };

  v8f acc[4][2];
  #pragma unroll
  for (int i = 0; i < 4; ++i)
    #pragma unroll
    for (int j = 0; j < 2; ++j) {
      v8f z = {0.f, 0.f, 0.f, 0.f, 0.f, 0.f, 0.f, 0.f};
      acc[i][j] = z;
    }

  const int NK = K / TK;
  issue_tile(0, 0);

  const int rsel = lane & 15;
  const int kA   = (lane < 16) ? 0 : 16;   // A frag: k in {kA..kA+15}
  const int kB   = (lane < 16) ? 0 : 16;   // B frag: k in {kB..kB+15}

  for (int kt = 0; kt < NK; ++kt) {
    const int buf = kt & 1;
    if (kt + 1 < NK) {
      issue_tile(kt + 1, buf ^ 1);
      asm volatile("s_wait_asynccnt 4" ::: "memory");  // this tile's 4 done
    } else {
      asm volatile("s_wait_asynccnt 0" ::: "memory");
    }
    __syncthreads();   // all waves' copies for `buf` complete

    v16bf bfr[2];
    #pragma unroll
    for (int j = 0; j < 2; ++j) {
      const __bf16* p = &Bl[buf][(wn * 32 + j * 16 + rsel) * PITCH + kB];
      bfr[j] = ld_frag2(p, p + 8);
    }
    #pragma unroll
    for (int i = 0; i < 4; ++i) {
      const __bf16* p = &Al[buf][(wm * 64 + i * 16 + rsel) * PITCH + kA];
      v16bf afr = ld_frag2(p, p + 8);
      #pragma unroll
      for (int j = 0; j < 2; ++j)
        acc[i][j] = __builtin_amdgcn_wmma_f32_16x16x32_bf16(
            false, afr, false, bfr[j], (short)0, acc[i][j], false, false);
    }
    __syncthreads();   // done reading `buf` before it is refilled
  }

  // epilogue: C/D layout — VGPR r <-> M row r + 8*(lane>=16), N = lane&15
  const int mh = (lane < 16) ? 0 : 8;
  #pragma unroll
  for (int i = 0; i < 4; ++i)
    #pragma unroll
    for (int j = 0; j < 2; ++j)
      #pragma unroll
      for (int r = 0; r < 8; ++r)
        C[(size_t)(m0 + wm * 64 + i * 16 + mh + r) * N
          + (n0 + wn * 32 + j * 16 + rsel)] = acc[i][j][r];
}

// ---------------------------------------------------------------------------
// 4) Fused hyperbolic GRU pointwise chain: one 256-thread block per row.
// ---------------------------------------------------------------------------
#define EPS_ 1e-5f

__device__ __forceinline__ float artanh_(float v) {
  v = fminf(fmaxf(v, -1.0f + EPS_), 1.0f - EPS_);
  return 0.5f * (log1pf(v) - log1pf(-v));
}
__device__ __forceinline__ float safenorm2(float s2) {
  return sqrtf(fmaxf(s2, 1e-15f));
}
__device__ __forceinline__ float sigmoid_(float v) {
  return 1.0f / (1.0f + expf(-v));
}

__device__ __forceinline__ float rsum(float v, float* sb)
{
  #pragma unroll
  for (int o = 16; o > 0; o >>= 1) v += __shfl_xor(v, o, 32);
  const int wid = threadIdx.x >> 5, lane = threadIdx.x & 31;
  if (lane == 0) sb[wid] = v;
  __syncthreads();
  if (wid == 0) {
    float s = (lane < 8) ? sb[lane] : 0.0f;
    #pragma unroll
    for (int o = 4; o > 0; o >>= 1) s += __shfl_xor(s, o, 32);
    if (lane == 0) sb[0] = s;
  }
  __syncthreads();
  float r = sb[0];
  __syncthreads();
  return r;
}

// gate = sigmoid(log0(mobius_add(matvec(x,Wx), mobius_add(matvec(h,Wh), b))))
__device__ __forceinline__ void gate_branch(
    const float* xw, float sxw, const float* hw, float shw, const float* bv,
    float x_n, float atx, float h_n, float ath, float* sb, float* g)
{
  const float wnx = safenorm2(sxw);
  const float s_a = tanhf(wnx / x_n * atx) / wnx;        // A = s_a * xw
  const float a2  = s_a * s_a * sxw;                     // ||A||^2
  const float wnh = safenorm2(shw);
  const float s_b = tanhf(wnh / h_n * ath) / wnh;        // Bm = s_b * hw
  const float b2  = s_b * s_b * shw;

  float d1 = 0.f, nb = 0.f;
  #pragma unroll
  for (int e = 0; e < 4; ++e) { d1 += hw[e] * bv[e]; nb += bv[e] * bv[e]; }
  d1 = rsum(d1, sb); nb = rsum(nb, sb);

  float xy  = s_b * d1;
  float den = 1.0f + 2.0f * xy + b2 * nb;
  float c1  = (1.0f + 2.0f * xy + nb) / den;
  float c2  = (1.0f - b2) / den;

  float Bv[4], d2 = 0.f, n2 = 0.f;
  #pragma unroll
  for (int e = 0; e < 4; ++e) {
    Bv[e] = c1 * (s_b * hw[e]) + c2 * bv[e];
    d2 += (s_a * xw[e]) * Bv[e];
    n2 += Bv[e] * Bv[e];
  }
  d2 = rsum(d2, sb); n2 = rsum(n2, sb);

  den = 1.0f + 2.0f * d2 + a2 * n2;
  c1  = (1.0f + 2.0f * d2 + n2) / den;
  c2  = (1.0f - a2) / den;

  float Zp[4], s2 = 0.f;
  #pragma unroll
  for (int e = 0; e < 4; ++e) {
    Zp[e] = c1 * (s_a * xw[e]) + c2 * Bv[e];
    s2 += Zp[e] * Zp[e];
  }
  s2 = rsum(s2, sb);
  const float zn = safenorm2(s2);
  const float gs = artanh_(zn) / zn;
  #pragma unroll
  for (int e = 0; e < 4; ++e) g[e] = sigmoid_(gs * Zp[e]);
}

__global__ __launch_bounds__(256)
void hyper_gru_pointwise(const float* __restrict__ inp, const float* __restrict__ prevh,
                         const float* __restrict__ bz, const float* __restrict__ br,
                         const float* __restrict__ bh,
                         const float* __restrict__ Gxz, const float* __restrict__ Ghz,
                         const float* __restrict__ Gxr, const float* __restrict__ Ghr,
                         const float* __restrict__ Gxh, const float* __restrict__ Ghh,
                         float* __restrict__ out)
{
  __shared__ float sb[8];
  const int t = threadIdx.x;
  const size_t base = (size_t)blockIdx.x * N_DIM;

  float h[4], xz[4], hz[4], xr[4], hr[4], xh[4], hh[4], vbz[4], vbr[4], vbh[4];
  float sx2 = 0, sh2 = 0, sxz = 0, shz = 0, sxr = 0, shr = 0, sxh = 0, shh = 0;
  #pragma unroll
  for (int e = 0; e < 4; ++e) {
    const int i = t + 256 * e;
    float xv = inp[base + i];   sx2 += xv * xv;
    h[e]  = prevh[base + i];    sh2 += h[e] * h[e];
    xz[e] = Gxz[base + i];      sxz += xz[e] * xz[e];
    hz[e] = Ghz[base + i];      shz += hz[e] * hz[e];
    xr[e] = Gxr[base + i];      sxr += xr[e] * xr[e];
    hr[e] = Ghr[base + i];      shr += hr[e] * hr[e];
    xh[e] = Gxh[base + i];      sxh += xh[e] * xh[e];
    hh[e] = Ghh[base + i];      shh += hh[e] * hh[e];
    vbz[e] = bz[i]; vbr[e] = br[i]; vbh[e] = bh[i];
  }
  sx2 = rsum(sx2, sb); sh2 = rsum(sh2, sb);
  sxz = rsum(sxz, sb); shz = rsum(shz, sb);
  sxr = rsum(sxr, sb); shr = rsum(shr, sb);
  sxh = rsum(sxh, sb); shh = rsum(shh, sb);

  const float x_n = safenorm2(sx2), h_n = safenorm2(sh2);
  const float atx = artanh_(x_n),   ath = artanh_(h_n);

  float z[4], r[4];
  gate_branch(xz, sxz, hz, shz, vbz, x_n, atx, h_n, ath, sb, z);
  gate_branch(xr, sxr, hr, shr, vbr, x_n, atx, h_n, ath, sb, r);

  // HL = mobius_add(matvec(h, Whh), bh)
  const float wnhh = safenorm2(shh);
  const float s_bh = tanhf(wnhh / h_n * ath) / wnhh;
  const float b2   = s_bh * s_bh * shh;
  float d1 = 0.f, nb = 0.f;
  #pragma unroll
  for (int e = 0; e < 4; ++e) { d1 += hh[e] * vbh[e]; nb += vbh[e] * vbh[e]; }
  d1 = rsum(d1, sb); nb = rsum(nb, sb);
  float xy  = s_bh * d1;
  float den = 1.0f + 2.0f * xy + b2 * nb;
  float c1  = (1.0f + 2.0f * xy + nb) / den;
  float c2  = (1.0f - b2) / den;
  float HL[4], hl2 = 0.f;
  #pragma unroll
  for (int e = 0; e < 4; ++e) {
    HL[e] = c1 * (s_bh * hh[e]) + c2 * vbh[e];
    hl2 += HL[e] * HL[e];
  }
  hl2 = rsum(hl2, sb);

  // p = mobius_pointwise_prod(HL, r)
  float ux[4], ux2 = 0.f;
  #pragma unroll
  for (int e = 0; e < 4; ++e) { ux[e] = r[e] * HL[e]; ux2 += ux[e] * ux[e]; }
  ux2 = rsum(ux2, sb);
  const float hln = safenorm2(hl2), uxn = safenorm2(ux2);
  const float t_p = tanhf(uxn / hln * artanh_(hln));
  const float s_p = t_p / uxn, p2 = t_p * t_p;

  // A1h = matvec(x, Wxh)
  const float wnxh = safenorm2(sxh);
  const float s_ah = tanhf(wnxh / x_n * atx) / wnxh;
  const float a2h  = s_ah * s_ah * sxh;

  // T = mobius_add(p, A1h)
  float dpa = 0.f;
  #pragma unroll
  for (int e = 0; e < 4; ++e) dpa += (s_p * ux[e]) * (s_ah * xh[e]);
  dpa = rsum(dpa, sb);
  den = 1.0f + 2.0f * dpa + p2 * a2h;
  c1  = (1.0f + 2.0f * dpa + a2h) / den;
  c2  = (1.0f - p2) / den;
  float T[4], T2 = 0.f;
  #pragma unroll
  for (int e = 0; e < 4; ++e) {
    T[e] = c1 * (s_p * ux[e]) + c2 * (s_ah * xh[e]);
    T2 += T[e] * T[e];
  }
  T2 = rsum(T2, sb);

  // MH = mobius_add(-h, T)
  float dht = 0.f;
  #pragma unroll
  for (int e = 0; e < 4; ++e) dht += h[e] * T[e];
  dht = rsum(dht, sb);
  xy  = -dht;
  den = 1.0f + 2.0f * xy + sh2 * T2;
  c1  = (1.0f + 2.0f * xy + T2) / den;
  c2  = (1.0f - sh2) / den;
  float MH[4], mh2 = 0.f;
  #pragma unroll
  for (int e = 0; e < 4; ++e) {
    MH[e] = -c1 * h[e] + c2 * T[e];
    mh2 += MH[e] * MH[e];
  }
  mh2 = rsum(mh2, sb);

  // q = mobius_pointwise_prod(MH, z)
  float uq[4], uq2 = 0.f;
  #pragma unroll
  for (int e = 0; e < 4; ++e) { uq[e] = z[e] * MH[e]; uq2 += uq[e] * uq[e]; }
  uq2 = rsum(uq2, sb);
  const float mhn = safenorm2(mh2), uqn = safenorm2(uq2);
  const float t_q = tanhf(uqn / mhn * artanh_(mhn));
  const float s_q = t_q / uqn, q2 = t_q * t_q;

  // h_next = mobius_add(h, q)
  float dhq = 0.f;
  #pragma unroll
  for (int e = 0; e < 4; ++e) dhq += h[e] * (s_q * uq[e]);
  dhq = rsum(dhq, sb);
  den = 1.0f + 2.0f * dhq + sh2 * q2;
  c1  = (1.0f + 2.0f * dhq + q2) / den;
  c2  = (1.0f - sh2) / den;
  #pragma unroll
  for (int e = 0; e < 4; ++e) {
    const int i = t + 256 * e;
    out[base + i] = c1 * h[e] + c2 * (s_q * uq[e]);
  }
}

// ---------------------------------------------------------------------------
// launch
// ---------------------------------------------------------------------------
extern "C" void kernel_launch(void* const* d_in, const int* in_sizes, int n_in,
                              void* d_out, int out_size, void* d_ws, size_t ws_size,
                              hipStream_t stream)
{
  const float* inp     = (const float*)d_in[0];
  const float* prevh   = (const float*)d_in[1];
  const float* w_inp_z = (const float*)d_in[2];
  const float* w_hid_z = (const float*)d_in[3];
  const float* b_hid_z = (const float*)d_in[4];
  const float* w_inp_r = (const float*)d_in[5];
  const float* w_hid_r = (const float*)d_in[6];
  const float* b_hid_r = (const float*)d_in[7];
  const float* w_inp_h = (const float*)d_in[8];
  const float* w_hid_h = (const float*)d_in[9];
  const float* b_hid_h = (const float*)d_in[10];
  float* out = (float*)d_out;

  // workspace layout: Ax, Ah (bf16), Wt[6] (bf16, transposed), G[6] (f32)
  __bf16* Ax = (__bf16*)d_ws;
  __bf16* Ah = Ax + (size_t)B_DIM * K_DIM;
  __bf16* Wt = Ah + (size_t)B_DIM * K_DIM;
  const size_t wsz = (size_t)N_DIM * K_DIM;
  float* G = (float*)(Wt + 6 * wsz);
  const size_t gsz = (size_t)B_DIM * N_DIM;

  const int n4 = B_DIM * K_DIM / 4;
  cast_f32_bf16<<<n4 / 256, 256, 0, stream>>>(inp,   Ax, n4);
  cast_f32_bf16<<<n4 / 256, 256, 0, stream>>>(prevh, Ah, n4);

  const float* Ws[6] = {w_inp_z, w_hid_z, w_inp_r, w_hid_r, w_inp_h, w_hid_h};
  dim3 tg(N_DIM / 32, K_DIM / 32);
  for (int i = 0; i < 6; ++i)
    transpose_cast<<<tg, 256, 0, stream>>>(Ws[i], Wt + i * wsz, K_DIM, N_DIM);

  dim3 gg(N_DIM / TN, B_DIM / TM);
  for (int i = 0; i < 6; ++i) {
    const __bf16* Am = (i & 1) ? Ah : Ax;   // even: inp-side, odd: hid-side
    gemm_bf16_wmma<<<gg, 256, 0, stream>>>(Am, Wt + i * wsz, G + i * gsz,
                                           B_DIM, N_DIM, K_DIM);
  }

  hyper_gru_pointwise<<<B_DIM, 256, 0, stream>>>(
      inp, prevh, b_hid_z, b_hid_r, b_hid_h,
      G + 0 * gsz, G + 1 * gsz, G + 2 * gsz, G + 3 * gsz, G + 4 * gsz, G + 5 * gsz,
      out);
}
